// MSAOuterProductMean_66984309948700
// MI455X (gfx1250) — matrix operand order repaired and checked by reading, not compiled, not run on hardware
//
#include <hip/hip_runtime.h>
#include <hip/hip_bf16.h>

// ---------------- types ----------------
typedef __bf16 v8bf  __attribute__((ext_vector_type(8)));
typedef __bf16 v16bf __attribute__((ext_vector_type(16)));
typedef float  v8f   __attribute__((ext_vector_type(8)));

static __device__ inline v16bf cat16(v8bf lo, v8bf hi) {
  return __builtin_shufflevector(lo, hi, 0,1,2,3,4,5,6,7,8,9,10,11,12,13,14,15);
}

#define N_SEQ    128
#define SEQ_LEN  256
#define MSA_DIM  256
#define HEAD_DIM 32
#define PAIR_DIM 128

// ---------------- kernel 0: w3 fp32 -> bf16 ----------------
__global__ __launch_bounds__(256) void cvt_w3_kernel(const float* __restrict__ w3,
                                                     __bf16* __restrict__ w3b) {
  int i = blockIdx.x * 256 + threadIdx.x;   // 131072 elems, grid covers exactly
  w3b[i] = (__bf16)w3[i];
}

// ---------------- kernel 1: LayerNorm + both projections ----------------
// One wave (32 lanes) per (n,pos) row. Lane l holds channels c = l*8+u.
// Outputs:
//   paT[(pos*32 + a)*128 + n] = proj_a   (A-tile layout [M=(i,a), K=n], bf16)
//   pbN[((n*256)+pos)*32 + a] = proj_b   (B-tile layout [K=n, N=(j,b)], bf16)
__global__ __launch_bounds__(256) void ln_proj_kernel(
    const float* __restrict__ msa, const float* __restrict__ nw,
    const float* __restrict__ nb,  const float* __restrict__ w1,
    const float* __restrict__ b1,  const float* __restrict__ w2,
    const float* __restrict__ b2,  __bf16* __restrict__ paT,
    __bf16* __restrict__ pbN) {
  const int wave = threadIdx.x >> 5;
  const int lane = threadIdx.x & 31;
  const int r    = blockIdx.x * 8 + wave;     // 0 .. 32767
  const int n    = r >> 8;
  const int pos  = r & 255;

  const float* xr = msa + (size_t)r * MSA_DIM + lane * 8;
  float x[8];
#pragma unroll
  for (int u = 0; u < 8; ++u) x[u] = xr[u];

  float s = 0.f, sq = 0.f;
#pragma unroll
  for (int u = 0; u < 8; ++u) { s += x[u]; sq += x[u] * x[u]; }
#pragma unroll
  for (int m = 16; m > 0; m >>= 1) {
    s  += __shfl_xor(s,  m, 32);
    sq += __shfl_xor(sq, m, 32);
  }
  const float mu   = s * (1.0f / MSA_DIM);
  const float var  = sq * (1.0f / MSA_DIM) - mu * mu;
  const float rstd = rsqrtf(var + 1e-5f);

  float xn[8];
#pragma unroll
  for (int u = 0; u < 8; ++u) {
    int c = lane * 8 + u;
    xn[u] = (x[u] - mu) * rstd * nw[c] + nb[c];
  }

  // Projections: lane `a` accumulates over all 256 channels (wave broadcast).
  float accA = 0.f, accB = 0.f;   // accA uses w2 (proj_a), accB uses w1 (proj_b)
  for (int cb = 0; cb < 32; ++cb) {
#pragma unroll
    for (int u = 0; u < 8; ++u) {
      float xc = __shfl(xn[u], cb, 32);
      int   c  = cb * 8 + u;
      accA = fmaf(xc, w2[c * HEAD_DIM + lane], accA);
      accB = fmaf(xc, w1[c * HEAD_DIM + lane], accB);
    }
  }
  accA += b2[lane];
  accB += b1[lane];

  paT[(size_t)(pos * HEAD_DIM + lane) * N_SEQ + n] = (__bf16)accA;
  pbN[(size_t)r * HEAD_DIM + lane]                 = (__bf16)accB;
}

// ---------------- kernel 2: fused outer-product-mean + w3 contraction ----------------
// Block (bx,by): i0 = by*4, j0 = bx*4.  256 threads = 8 waves.
// Stage 1: C[128x128] = A[128x128] * B[128x128] over K=n (wave w owns M-strip w).
// Stage 2: out-tile[16 pairs x 128] = E[16x1024] * w3b[1024x128] (wave w owns 16 cols).
#define LA_S 136   // padded row stride (elems) for A tile [128 x 128]
#define LB_S 144   // padded row stride for B tile [128 x 128]
#define LE_S 1032  // padded row stride for E [16 x 1024]

__global__ __launch_bounds__(256) void fused_outer_kernel(
    const __bf16* __restrict__ paT, const __bf16* __restrict__ pbN,
    const __bf16* __restrict__ w3b, const float* __restrict__ b3,
    float* __restrict__ out) {
  __shared__ __attribute__((aligned(32))) char smem[(128 * LA_S + 128 * LB_S) * 2];
  __bf16* lA = (__bf16*)smem;              // [128][LA_S]  (M=(il,a), K=n)
  __bf16* lB = lA + 128 * LA_S;            // [128][LB_S]  (K=n, N=(jl,b))
  __bf16* lE = (__bf16*)smem;              // [16][LE_S], aliases lA/lB after barrier

  const int tid  = threadIdx.x;
  const int wave = tid >> 5;
  const int lane = tid & 31;
  const int hi   = lane >> 4;
  const int i0   = blockIdx.y * 4;
  const int j0   = blockIdx.x * 4;

  // --- load A tile: contiguous 32KB chunk of paT rows [i0*32, i0*32+128) ---
  {
    const uint4* src = (const uint4*)(paT + (size_t)(i0 * HEAD_DIM) * N_SEQ);
#pragma unroll
    for (int p = 0; p < 8; ++p) {
      int idx = p * 256 + tid;             // uint4 = 8 bf16; 16 per 128-el row
      int row = idx >> 4, col = (idx & 15) << 3;
      *(uint4*)(lA + row * LA_S + col) = src[idx];
    }
  }
  // --- load B tile: rows n=0..127, 128 bf16 each, from pbN[(n*256+j0)*32] ---
  {
#pragma unroll
    for (int p = 0; p < 8; ++p) {
      int idx = p * 256 + tid;
      int row = idx >> 4, col = (idx & 15) << 3;
      const uint4* src =
          (const uint4*)(pbN + ((size_t)row * SEQ_LEN + j0) * HEAD_DIM + col);
      *(uint4*)(lB + row * LB_S + col) = *src;
    }
  }
  __syncthreads();

  // ---------------- stage 1: 128x128x128 GEMM via WMMA bf16 ----------------
  v8f acc[8];
#pragma unroll
  for (int t = 0; t < 8; ++t) acc[t] = (v8f){0.f,0.f,0.f,0.f,0.f,0.f,0.f,0.f};

  const int mrow = wave * 16 + (lane & 15);
#pragma unroll
  for (int ks = 0; ks < 4; ++ks) {
    const int k0 = ks * 32;
    // A fragment: lane=M, two 8-elem K chunks per half-lane group
    v8bf a_lo = *(const v8bf*)(lA + mrow * LA_S + k0 + 8 * hi);
    v8bf a_hi = *(const v8bf*)(lA + mrow * LA_S + k0 + 16 + 8 * hi);
    v16bf af  = cat16(a_lo, a_hi);
#pragma unroll
    for (int nt = 0; nt < 8; ++nt) {
      // B fragment: lane=N column, 16 K elems in table order
      v16bf bf;
#pragma unroll
      for (int s = 0; s < 16; ++s)
        bf[s] = lB[(k0 + (s < 8 ? 8 * hi + s : 16 + 8 * hi + (s - 8))) * LB_S + nt * 16 + (lane & 15)];
      acc[nt] = __builtin_amdgcn_wmma_f32_16x16x32_bf16(
          false, af, false, bf, (short)0, acc[nt], false, false);
    }
  }
  __syncthreads();   // everyone done reading lA/lB before E overwrites them

  // scatter C -> E[pair = il*4+jl][a*32+b], scaled by 1/n_seq, as bf16
#pragma unroll
  for (int nt = 0; nt < 8; ++nt) {
#pragma unroll
    for (int v = 0; v < 8; ++v) {
      int m  = wave * 16 + v + 8 * hi;   // (il,a)
      int nn = nt * 16 + (lane & 15);    // (jl,b)
      int pair = (m >> 5) * 4 + (nn >> 5);
      lE[pair * LE_S + (m & 31) * HEAD_DIM + (nn & 31)] =
          (__bf16)(acc[nt][v] * (1.0f / N_SEQ));
    }
  }
  __syncthreads();

  // ---------------- stage 2: [16 x 1024] x [1024 x 128] via WMMA bf16 ------
  {
    v8f acc2 = (v8f){0.f,0.f,0.f,0.f,0.f,0.f,0.f,0.f};
    const int p0   = wave * 16;
    const int erow = lane & 15;
    for (int ks = 0; ks < 32; ++ks) {
      const int k0 = ks * 32;
      v8bf e_lo = *(const v8bf*)(lE + erow * LE_S + k0 + 8 * hi);
      v8bf e_hi = *(const v8bf*)(lE + erow * LE_S + k0 + 16 + 8 * hi);
      v16bf ef  = cat16(e_lo, e_hi);
      v16bf wf;
#pragma unroll
      for (int s = 0; s < 16; ++s)
        wf[s] = w3b[(size_t)(k0 + (s < 8 ? 8 * hi + s : 16 + 8 * hi + (s - 8))) * PAIR_DIM + p0 + (lane & 15)];
      if (ks < 31)  // hint next w3 slab into cache (global_prefetch_b8)
        __builtin_prefetch(w3b + (size_t)(k0 + 32 + lane) * PAIR_DIM + p0, 0, 3);
      acc2 = __builtin_amdgcn_wmma_f32_16x16x32_bf16(
          false, ef, false, wf, (short)0, acc2, false, false);
    }
    const int p = p0 + (lane & 15);
    const float bias = b3[p];
#pragma unroll
    for (int v = 0; v < 8; ++v) {
      int pair = v + 8 * hi;             // D row
      int il = pair >> 2, jl = pair & 3;
      size_t o = (((size_t)(i0 + il)) * SEQ_LEN + (j0 + jl)) * PAIR_DIM + p;
      out[o] = acc2[v] + bias;
    }
  }
}

// ---------------- host launch ----------------
extern "C" void kernel_launch(void* const* d_in, const int* in_sizes, int n_in,
                              void* d_out, int out_size, void* d_ws, size_t ws_size,
                              hipStream_t stream) {
  const float* msa    = (const float*)d_in[0];
  const float* norm_w = (const float*)d_in[1];
  const float* norm_b = (const float*)d_in[2];
  const float* w1     = (const float*)d_in[3];
  const float* b1     = (const float*)d_in[4];
  const float* w2     = (const float*)d_in[5];
  const float* b2     = (const float*)d_in[6];
  const float* w3     = (const float*)d_in[7];
  const float* b3     = (const float*)d_in[8];
  float* out = (float*)d_out;

  __bf16* paT = (__bf16*)d_ws;                       // [256*32, 128]
  __bf16* pbN = paT + (size_t)SEQ_LEN * HEAD_DIM * N_SEQ;   // [128*256, 32]
  __bf16* w3b = pbN + (size_t)N_SEQ * SEQ_LEN * HEAD_DIM;   // [1024, 128]

  cvt_w3_kernel<<<512, 256, 0, stream>>>(w3, w3b);
  ln_proj_kernel<<<(N_SEQ * SEQ_LEN) / 8, 256, 0, stream>>>(
      msa, norm_w, norm_b, w1, b1, w2, b2, paT, pbN);
  fused_outer_kernel<<<dim3(SEQ_LEN / 4, SEQ_LEN / 4), 256, 0, stream>>>(
      paT, pbN, w3b, b3, out);
}
